// CounterfactualRepairAttention_59648505806981
// MI455X (gfx1250) — compile-verified, matrix operand order checked by reading, not run on hardware
//
#include <hip/hip_runtime.h>
#include <hip/hip_bf16.h>
#include <math.h>

// ---------------------------------------------------------------------------
// CounterfactualRepairAttention for MI455X (gfx1250, wave32, WMMA bf16->f32)
// B=8, L=1024, D=768
// Register-blocked WMMA GEMMs: proj waves do 32x64 tiles (8 WMMA / 6 frag
// loads per k-step), score waves do 16x32 per score type (6 WMMA / 9 loads).
// ---------------------------------------------------------------------------

#define B_ 8
#define L_ 1024
#define D_ 768

typedef __attribute__((ext_vector_type(16))) __bf16 v16bf;
typedef __attribute__((ext_vector_type(8)))  __bf16 v8bf;
typedef __attribute__((ext_vector_type(8)))  float  v8f;

struct Ptr6 { const float* p[6]; };

// float -> bf16 (round to nearest even), stored as raw u16
__device__ __forceinline__ unsigned short f2bf(float f) {
  union { float f; unsigned int u; } v; v.f = f;
  unsigned int u = v.u;
  unsigned int r = u + 0x7FFFu + ((u >> 16) & 1u);
  return (unsigned short)(r >> 16);
}

// Load a WMMA 16-bit operand fragment from a row-major bf16 buffer.
// CDNA5 16-bit layout: lane half h holds K in {8h..8h+7} (elems 0..7) and
// {16+8h..16+8h+7} (elems 8..15)  ->  two contiguous 16B loads (merge to b128).
__device__ __forceinline__ v16bf load_frag_row(const unsigned short* buf,
                                               int row, int ld, int k0, int h) {
  const __bf16* base = (const __bf16*)(buf + (size_t)row * ld + k0);
  v8bf lo = *(const v8bf*)(base + 8 * h);
  v8bf hi = *(const v8bf*)(base + 16 + 8 * h);
  v16bf r;
#pragma unroll
  for (int i = 0; i < 8; ++i) { r[i] = lo[i]; r[i + 8] = hi[i]; }
  return r;
}

// ---------------------------------------------------------------------------
// 1) sep_pos per batch
// ---------------------------------------------------------------------------
__global__ void k_seppos(const int* __restrict__ x_ids,
                         const int* __restrict__ pad_ptr,
                         int* __restrict__ sep) {
  int b = blockIdx.x;
  int pad = pad_ptr[0];
  __shared__ int red[256];
  int c = 0;
  for (int l = threadIdx.x; l < L_; l += 256) c += (x_ids[b * L_ + l] != pad);
  red[threadIdx.x] = c; __syncthreads();
  for (int s = 128; s > 0; s >>= 1) {
    if (threadIdx.x < s) red[threadIdx.x] += red[threadIdx.x + s];
    __syncthreads();
  }
  if (threadIdx.x == 0) {
    int v = red[0] / 2;
    if (v < 1) v = 1;
    if (v > L_ - 2) v = L_ - 2;
    sep[b] = v;
  }
}

// ---------------------------------------------------------------------------
// 2) x -> bf16
// ---------------------------------------------------------------------------
__global__ void k_cvt_x(const float* __restrict__ x, unsigned short* __restrict__ xb) {
  size_t i = (size_t)blockIdx.x * 256 + threadIdx.x;
  xb[i] = f2bf(x[i]);
}

// ---------------------------------------------------------------------------
// 3) weight -> bf16, transposed: Wt[n][k] = W[k][n]
// ---------------------------------------------------------------------------
__global__ void k_w_cvt(Ptr6 ws, unsigned short* __restrict__ wt) {
  int widx = blockIdx.z;
  int idx = blockIdx.x * 256 + threadIdx.x;    // over 768*768
  int n = idx / D_, k = idx % D_;
  const float* W = ws.p[widx];
  wt[(size_t)widx * D_ * D_ + (size_t)n * D_ + k] = f2bf(W[(size_t)k * D_ + n]);
}

// ---------------------------------------------------------------------------
// 4) anomaly logits: alog = x @ Wa + ba  (one wave per row)
// ---------------------------------------------------------------------------
__global__ void k_logits(const float* __restrict__ x, const float* __restrict__ Wa,
                         const float* __restrict__ ba, float* __restrict__ alog) {
  int wid = threadIdx.x >> 5, lane = threadIdx.x & 31;
  int row = blockIdx.x * 8 + wid;
  float s = 0.f;
  for (int i = lane; i < D_; i += 32) s += x[(size_t)row * D_ + i] * Wa[i];
  for (int m = 16; m > 0; m >>= 1) s += __shfl_xor(s, m, 32);
  if (lane == 0) alog[row] = s + ba[0];
}

// ---------------------------------------------------------------------------
// 5) gate = masked softmax over false segment (one block per batch)
// ---------------------------------------------------------------------------
__global__ void k_gate(const float* __restrict__ alog, const int* __restrict__ x_ids,
                       const int* __restrict__ pad_ptr, const int* __restrict__ sep,
                       float* __restrict__ gate) {
  int b = blockIdx.x;
  int pad = pad_ptr[0];
  int sp = sep[b];
  __shared__ float red[256];
  __shared__ float ev[L_];
  float mx = -3.0e38f;
  for (int l = threadIdx.x; l < L_; l += 256) {
    bool fm = (l < sp) && (x_ids[b * L_ + l] != pad);
    float v = fm ? alog[b * L_ + l] : -3.0e38f;
    ev[l] = v;
    mx = fmaxf(mx, v);
  }
  red[threadIdx.x] = mx; __syncthreads();
  for (int s = 128; s > 0; s >>= 1) {
    if (threadIdx.x < s) red[threadIdx.x] = fmaxf(red[threadIdx.x], red[threadIdx.x + s]);
    __syncthreads();
  }
  mx = red[0]; __syncthreads();
  float sm = 0.f;
  for (int l = threadIdx.x; l < L_; l += 256) {
    float v = ev[l];
    float e = (v > -1.0e37f) ? __expf(v - mx) : 0.0f;
    ev[l] = e;
    sm += e;
  }
  red[threadIdx.x] = sm; __syncthreads();
  for (int s = 128; s > 0; s >>= 1) {
    if (threadIdx.x < s) red[threadIdx.x] += red[threadIdx.x + s];
    __syncthreads();
  }
  float inv = 1.0f / fmaxf(red[0], 1e-8f);
  for (int l = threadIdx.x; l < L_; l += 256)
    gate[b * L_ + l] = ev[l] * inv;
}

// ---------------------------------------------------------------------------
// 6) projections: proj[w] = bf16(x @ W[w] + b[w]), via WMMA bf16 -> f32
//    Wave tile 32(M) x 64(N): 2 A-frags + 4 B-frags -> 8 WMMA per k-step.
//    Block = 8 waves (4 across M, 2 across N) -> 128x128 block tile.
//    grid: (768/128, 8192/128, 6)
// ---------------------------------------------------------------------------
__global__ void k_proj(const unsigned short* __restrict__ xb,
                       const unsigned short* __restrict__ wt,
                       Ptr6 biases,
                       unsigned short* __restrict__ proj) {
  int wsel = blockIdx.z;
  int wid = threadIdx.x >> 5, lane = threadIdx.x & 31;
  int wm = wid & 3, wn = wid >> 2;
  int h = lane >> 4, ln = lane & 15;
  int mbase = blockIdx.y * 128 + wm * 32;
  int nbase = blockIdx.x * 128 + wn * 64;
  const unsigned short* W = wt + (size_t)wsel * D_ * D_;

  int arow0 = mbase + ln;
  int arow1 = mbase + 16 + ln;

  v8f acc[2][4];
#pragma unroll
  for (int i = 0; i < 2; ++i)
#pragma unroll
    for (int j = 0; j < 4; ++j) acc[i][j] = (v8f){};

  for (int k0 = 0; k0 < D_; k0 += 32) {
    v16bf a0 = load_frag_row(xb, arow0, D_, k0, h);
    v16bf a1 = load_frag_row(xb, arow1, D_, k0, h);
#pragma unroll
    for (int j = 0; j < 4; ++j) {
      v16bf b = load_frag_row(W, nbase + j * 16 + ln, D_, k0, h);
      acc[0][j] = __builtin_amdgcn_wmma_f32_16x16x32_bf16(false, a0, false, b,
                                                          (short)0, acc[0][j], false, false);
      acc[1][j] = __builtin_amdgcn_wmma_f32_16x16x32_bf16(false, a1, false, b,
                                                          (short)0, acc[1][j], false, false);
    }
  }

  const float* bias = biases.p[wsel];
  unsigned short* out = proj + (size_t)wsel * (B_ * L_) * D_;
#pragma unroll
  for (int j = 0; j < 4; ++j) {
    int ncol = nbase + j * 16 + ln;
    float bv = bias[ncol];
#pragma unroll
    for (int i = 0; i < 2; ++i) {
#pragma unroll
      for (int r = 0; r < 8; ++r) {
        int mrow = mbase + i * 16 + 8 * h + r;
        out[(size_t)mrow * D_ + ncol] = f2bf(acc[i][j][r] + bv);
      }
    }
  }
}

// ---------------------------------------------------------------------------
// 7) scores: S_sup = (q_s k_s^T)*scale ; S2 = (q_r k_r^T)*scale + tanh((q_c k_c^T)*scale)
//    Wave tile 16(L) x 32(M) per score type: 3 A + 6 B frags -> 6 WMMA per k-step.
//    Block = 8 waves covering 256 m-columns; grid: (1024/256, 64, 8)
// ---------------------------------------------------------------------------
__global__ void k_scores(const unsigned short* __restrict__ proj,
                         float* __restrict__ ssup, float* __restrict__ s2) {
  int b = blockIdx.z;
  int ltile = blockIdx.y;
  int wid = threadIdx.x >> 5, lane = threadIdx.x & 31;
  int h = lane >> 4, ln = lane & 15;
  int mbase = blockIdx.x * 256 + wid * 32;

  const size_t pstride = (size_t)(B_ * L_) * D_;
  const unsigned short* qs = proj + 0 * pstride;
  const unsigned short* ks = proj + 1 * pstride;
  const unsigned short* qc = proj + 2 * pstride;
  const unsigned short* kc = proj + 3 * pstride;
  const unsigned short* qr = proj + 4 * pstride;
  const unsigned short* kr = proj + 5 * pstride;

  int arow  = b * L_ + ltile * 16 + ln;
  int brow0 = b * L_ + mbase + ln;
  int brow1 = b * L_ + mbase + 16 + ln;

  v8f aS0 = {}, aS1 = {}, aC0 = {}, aC1 = {}, aR0 = {}, aR1 = {};
  for (int k0 = 0; k0 < D_; k0 += 32) {
    v16bf a0 = load_frag_row(qs, arow, D_, k0, h);
    v16bf b00 = load_frag_row(ks, brow0, D_, k0, h);
    v16bf b01 = load_frag_row(ks, brow1, D_, k0, h);
    aS0 = __builtin_amdgcn_wmma_f32_16x16x32_bf16(false, a0, false, b00, (short)0, aS0, false, false);
    aS1 = __builtin_amdgcn_wmma_f32_16x16x32_bf16(false, a0, false, b01, (short)0, aS1, false, false);

    v16bf a1 = load_frag_row(qc, arow, D_, k0, h);
    v16bf b10 = load_frag_row(kc, brow0, D_, k0, h);
    v16bf b11 = load_frag_row(kc, brow1, D_, k0, h);
    aC0 = __builtin_amdgcn_wmma_f32_16x16x32_bf16(false, a1, false, b10, (short)0, aC0, false, false);
    aC1 = __builtin_amdgcn_wmma_f32_16x16x32_bf16(false, a1, false, b11, (short)0, aC1, false, false);

    v16bf a2 = load_frag_row(qr, arow, D_, k0, h);
    v16bf b20 = load_frag_row(kr, brow0, D_, k0, h);
    v16bf b21 = load_frag_row(kr, brow1, D_, k0, h);
    aR0 = __builtin_amdgcn_wmma_f32_16x16x32_bf16(false, a2, false, b20, (short)0, aR0, false, false);
    aR1 = __builtin_amdgcn_wmma_f32_16x16x32_bf16(false, a2, false, b21, (short)0, aR1, false, false);
  }

  const float scale = 0.03608439182435161f; // 1/sqrt(768)
#pragma unroll
  for (int r = 0; r < 8; ++r) {
    int lrow = ltile * 16 + 8 * h + r;
    size_t o0 = ((size_t)(b * L_ + lrow)) * L_ + mbase + ln;
    size_t o1 = o0 + 16;
    ssup[o0] = aS0[r] * scale;
    ssup[o1] = aS1[r] * scale;
    s2[o0]   = aR0[r] * scale + tanhf(aC0[r] * scale);
    s2[o1]   = aR1[r] * scale + tanhf(aC1[r] * scale);
  }
}

// ---------------------------------------------------------------------------
// 8) zero c-vectors
// ---------------------------------------------------------------------------
__global__ void k_zero(float* __restrict__ p, int n) {
  int i = blockIdx.x * 256 + threadIdx.x;
  if (i < n) p[i] = 0.0f;
}

// ---------------------------------------------------------------------------
// 9) per-row masked softmax + gate-weighted accumulation into c_sup / c_rep
//    one block per (b,l) row; rows with gate==0 exit early (uniform)
// ---------------------------------------------------------------------------
__global__ void k_softrow(const float* __restrict__ ssup, const float* __restrict__ s2,
                          const float* __restrict__ gate,
                          const int* __restrict__ x_ids, const int* __restrict__ pad_ptr,
                          const int* __restrict__ sep,
                          float* __restrict__ csup, float* __restrict__ crep) {
  int row = blockIdx.x;
  int b = row >> 10;
  float g = gate[row];
  if (g == 0.0f) return;                    // uniform per block
  int pad = pad_ptr[0];
  int sp = sep[b];
  __shared__ float red[256];
  const float* S1 = ssup + (size_t)row * L_;
  const float* S2r = s2 + (size_t)row * L_;

  float v1[4], v2[4]; bool om[4];
  float m1 = -3.0e38f, m2 = -3.0e38f;
#pragma unroll
  for (int i = 0; i < 4; ++i) {
    int m = threadIdx.x + 256 * i;
    bool o = (m > sp) && (x_ids[b * L_ + m] != pad);
    om[i] = o;
    v1[i] = o ? S1[m] : -3.0e38f;
    v2[i] = o ? S2r[m] : -3.0e38f;
    m1 = fmaxf(m1, v1[i]);
    m2 = fmaxf(m2, v2[i]);
  }
  red[threadIdx.x] = m1; __syncthreads();
  for (int s = 128; s > 0; s >>= 1) {
    if (threadIdx.x < s) red[threadIdx.x] = fmaxf(red[threadIdx.x], red[threadIdx.x + s]);
    __syncthreads();
  }
  m1 = red[0]; __syncthreads();
  red[threadIdx.x] = m2; __syncthreads();
  for (int s = 128; s > 0; s >>= 1) {
    if (threadIdx.x < s) red[threadIdx.x] = fmaxf(red[threadIdx.x], red[threadIdx.x + s]);
    __syncthreads();
  }
  m2 = red[0]; __syncthreads();

  float s1 = 0.f, s2s = 0.f;
#pragma unroll
  for (int i = 0; i < 4; ++i) {
    float e1 = om[i] ? __expf(v1[i] - m1) : 0.0f;
    float e2 = om[i] ? __expf(v2[i] - m2) : 0.0f;
    v1[i] = e1; v2[i] = e2;
    s1 += e1; s2s += e2;
  }
  red[threadIdx.x] = s1; __syncthreads();
  for (int s = 128; s > 0; s >>= 1) {
    if (threadIdx.x < s) red[threadIdx.x] += red[threadIdx.x + s];
    __syncthreads();
  }
  s1 = red[0]; __syncthreads();
  red[threadIdx.x] = s2s; __syncthreads();
  for (int s = 128; s > 0; s >>= 1) {
    if (threadIdx.x < s) red[threadIdx.x] += red[threadIdx.x + s];
    __syncthreads();
  }
  s2s = red[0];

  float i1 = g / fmaxf(s1, 1e-30f);
  float i2 = g / fmaxf(s2s, 1e-30f);
#pragma unroll
  for (int i = 0; i < 4; ++i) {
    int m = threadIdx.x + 256 * i;
    if (om[i]) {
      atomicAdd(&csup[b * L_ + m], v1[i] * i1);
      atomicAdd(&crep[b * L_ + m], v2[i] * i2);
    }
  }
}

// ---------------------------------------------------------------------------
// 10) weighted sums over x: fused_in[b] = [gate@x | c_rep@x | c_sup@x]
// ---------------------------------------------------------------------------
__global__ void k_weighted(const float* __restrict__ x, const float* __restrict__ gate,
                           const float* __restrict__ crep, const float* __restrict__ csup,
                           float* __restrict__ fusedin) {
  int b = blockIdx.z, which = blockIdx.y;
  int d = blockIdx.x * 256 + threadIdx.x;
  const float* coef = (which == 0) ? gate : ((which == 1) ? crep : csup);
  float acc = 0.f;
  for (int m = 0; m < L_; ++m)
    acc += coef[b * L_ + m] * x[((size_t)(b * L_ + m)) * D_ + d];
  fusedin[b * (3 * D_) + which * D_ + d] = acc;
}

// ---------------------------------------------------------------------------
// 11) MLP layers + layernorm (tiny: B=8)
// ---------------------------------------------------------------------------
__global__ void k_ffn1(const float* __restrict__ fusedin, const float* __restrict__ Wf1,
                       const float* __restrict__ bf1, float* __restrict__ h1) {
  int b = blockIdx.z;
  int j = blockIdx.x * 256 + threadIdx.x;
  float s = bf1[j];
  for (int i = 0; i < 3 * D_; ++i)
    s += fusedin[b * (3 * D_) + i] * Wf1[(size_t)i * D_ + j];
  h1[b * D_ + j] = fmaxf(s, 0.0f);
}

__global__ void k_ffn2(const float* __restrict__ h1, const float* __restrict__ Wf2,
                       const float* __restrict__ bf2, float* __restrict__ pre) {
  int b = blockIdx.z;
  int j = blockIdx.x * 256 + threadIdx.x;
  float s = bf2[j];
  for (int i = 0; i < D_; ++i)
    s += h1[b * D_ + i] * Wf2[(size_t)i * D_ + j];
  pre[b * D_ + j] = s;
}

__global__ void k_ln(const float* __restrict__ pre, const float* __restrict__ gamma,
                     const float* __restrict__ beta, float* __restrict__ out) {
  int b = blockIdx.x;
  __shared__ float red[256];
  float s = 0.f;
  for (int j = threadIdx.x; j < D_; j += 256) s += pre[b * D_ + j];
  red[threadIdx.x] = s; __syncthreads();
  for (int t = 128; t > 0; t >>= 1) {
    if (threadIdx.x < t) red[threadIdx.x] += red[threadIdx.x + t];
    __syncthreads();
  }
  float mu = red[0] / (float)D_; __syncthreads();
  float v = 0.f;
  for (int j = threadIdx.x; j < D_; j += 256) {
    float d = pre[b * D_ + j] - mu;
    v += d * d;
  }
  red[threadIdx.x] = v; __syncthreads();
  for (int t = 128; t > 0; t >>= 1) {
    if (threadIdx.x < t) red[threadIdx.x] += red[threadIdx.x + t];
    __syncthreads();
  }
  float inv = rsqrtf(red[0] / (float)D_ + 1e-5f);
  for (int j = threadIdx.x; j < D_; j += 256)
    out[b * D_ + j] = (pre[b * D_ + j] - mu) * inv * gamma[j] + beta[j];
}

// ---------------------------------------------------------------------------
// host launch
// ---------------------------------------------------------------------------
extern "C" void kernel_launch(void* const* d_in, const int* in_sizes, int n_in,
                              void* d_out, int out_size, void* d_ws, size_t ws_size,
                              hipStream_t stream) {
  const float* x      = (const float*)d_in[0];
  const int*   x_ids  = (const int*)d_in[1];
  const int*   pad    = (const int*)d_in[2];
  const float* Wa     = (const float*)d_in[3];
  const float* ba     = (const float*)d_in[4];
  const float* Wqs = (const float*)d_in[5];  const float* bqs = (const float*)d_in[6];
  const float* Wks = (const float*)d_in[7];  const float* bks = (const float*)d_in[8];
  const float* Wqc = (const float*)d_in[9];  const float* bqc = (const float*)d_in[10];
  const float* Wkc = (const float*)d_in[11]; const float* bkc = (const float*)d_in[12];
  const float* Wqr = (const float*)d_in[13]; const float* bqr = (const float*)d_in[14];
  const float* Wkr = (const float*)d_in[15]; const float* bkr = (const float*)d_in[16];
  const float* Wf1 = (const float*)d_in[17]; const float* bf1 = (const float*)d_in[18];
  const float* Wf2 = (const float*)d_in[19]; const float* bf2 = (const float*)d_in[20];
  const float* gamma = (const float*)d_in[21];
  const float* beta  = (const float*)d_in[22];
  float* out = (float*)d_out;

  // workspace layout (256B aligned slots)
  char* ws = (char*)d_ws;
  size_t off = 0;
  auto alloc = [&](size_t bytes) {
    size_t o = off;
    off = (off + bytes + 255) & ~(size_t)255;
    return o;
  };
  const size_t ROWS = (size_t)B_ * L_;                        // 8192
  size_t oXB   = alloc(ROWS * D_ * 2);                        // x in bf16
  size_t oWT   = alloc(6 * (size_t)D_ * D_ * 2);              // 6 transposed bf16 weights
  size_t oPROJ = alloc(6 * ROWS * D_ * 2);                    // 6 bf16 projections
  size_t oSSUP = alloc((size_t)B_ * L_ * L_ * 4);             // sup scores f32
  size_t oS2   = alloc((size_t)B_ * L_ * L_ * 4);             // rep + tanh(con)
  size_t oALOG = alloc(ROWS * 4);
  size_t oGATE = alloc(ROWS * 4);
  size_t oSEP  = alloc(B_ * 4);
  size_t oCSUP = alloc((size_t)B_ * L_ * 4);                  // adjacent to CREP
  size_t oCREP = alloc((size_t)B_ * L_ * 4);
  size_t oFIN  = alloc((size_t)B_ * 3 * D_ * 4);
  size_t oH1   = alloc((size_t)B_ * D_ * 4);
  size_t oPRE  = alloc((size_t)B_ * D_ * 4);
  (void)ws_size; (void)n_in; (void)in_sizes; (void)out_size; (void)oCREP;

  unsigned short* XB   = (unsigned short*)(ws + oXB);
  unsigned short* WT   = (unsigned short*)(ws + oWT);
  unsigned short* PROJ = (unsigned short*)(ws + oPROJ);
  float* SSUP = (float*)(ws + oSSUP);
  float* S2   = (float*)(ws + oS2);
  float* ALOG = (float*)(ws + oALOG);
  float* GATE = (float*)(ws + oGATE);
  int*   SEP  = (int*)(ws + oSEP);
  float* CSUP = (float*)(ws + oCSUP);
  float* CREP = (float*)(ws + oCREP);
  float* FIN  = (float*)(ws + oFIN);
  float* H1   = (float*)(ws + oH1);
  float* PRE  = (float*)(ws + oPRE);

  Ptr6 wptrs; wptrs.p[0] = Wqs; wptrs.p[1] = Wks; wptrs.p[2] = Wqc;
  wptrs.p[3] = Wkc; wptrs.p[4] = Wqr; wptrs.p[5] = Wkr;
  Ptr6 bptrs; bptrs.p[0] = bqs; bptrs.p[1] = bks; bptrs.p[2] = bqc;
  bptrs.p[3] = bkc; bptrs.p[4] = bqr; bptrs.p[5] = bkr;

  k_seppos<<<B_, 256, 0, stream>>>(x_ids, pad, SEP);
  k_cvt_x<<<(int)(ROWS * D_ / 256), 256, 0, stream>>>(x, XB);
  k_w_cvt<<<dim3(D_ * D_ / 256, 1, 6), 256, 0, stream>>>(wptrs, WT);
  k_logits<<<(int)(ROWS / 8), 256, 0, stream>>>(x, Wa, ba, ALOG);
  k_gate<<<B_, 256, 0, stream>>>(ALOG, x_ids, pad, SEP, GATE);

  // 6 projections: block tile 128x128 (8 waves of 32x64); grid (6, 64, 6)
  k_proj<<<dim3(D_ / 128, (int)(ROWS / 128), 6), 256, 0, stream>>>(XB, WT, bptrs, PROJ);

  // scores: 8 waves x 32 m-cols = 256 m per block; grid (4, 64, 8)
  k_scores<<<dim3(L_ / 256, L_ / 16, B_), 256, 0, stream>>>(PROJ, SSUP, S2);

  // zero CSUP+CREP (contiguous: 2*8192 floats)
  k_zero<<<(2 * B_ * L_ + 255) / 256, 256, 0, stream>>>(CSUP, 2 * B_ * L_);

  k_softrow<<<(int)ROWS, 256, 0, stream>>>(SSUP, S2, GATE, x_ids, pad, SEP, CSUP, CREP);

  k_weighted<<<dim3(D_ / 256, 3, B_), 256, 0, stream>>>(x, GATE, CREP, CSUP, FIN);
  k_ffn1<<<dim3(D_ / 256, 1, B_), 256, 0, stream>>>(FIN, Wf1, bf1, H1);
  k_ffn2<<<dim3(D_ / 256, 1, B_), 256, 0, stream>>>(H1, Wf2, bf2, PRE);
  k_ln<<<B_, 256, 0, stream>>>(PRE, gamma, beta, out);
}
